// TreeNNBatch_84061099917532
// MI455X (gfx1250) — matrix operand order repaired and checked by reading, not compiled, not run on hardware
//
#include <hip/hip_runtime.h>
#include <hip/hip_bf16.h>

typedef _Float16 h16;
typedef __attribute__((ext_vector_type(16))) _Float16 v16h;
typedef __attribute__((ext_vector_type(8)))  _Float16 v8h;
typedef __attribute__((ext_vector_type(8)))  float    v8f;
typedef __attribute__((ext_vector_type(4)))  float    v4f;

// Problem constants (match reference)
#define BB   1024
#define DD   5
#define NTOT 64512
#define HID  128
#define REP  128
#define XW   1408   // 5*HID + 6*REP
#define BMK  1000
#define BMKP 1024   // padded to multiple of 32

// ---------------------------------------------------------------------------
// Fragment loaders: 8 contiguous elements -> half of a v16h, f32 converted.
// ---------------------------------------------------------------------------
__device__ inline void load8(const h16* __restrict__ row, int c, v16h& a, int base) {
  v8h t = *(const v8h*)(row + c);
#pragma unroll
  for (int i = 0; i < 8; ++i) a[base + i] = t[i];
}
__device__ inline void load8(const float* __restrict__ row, int c, v16h& a, int base) {
  v4f t0 = *(const v4f*)(row + c);
  v4f t1 = *(const v4f*)(row + c + 4);
#pragma unroll
  for (int i = 0; i < 4; ++i) { a[base + i] = (h16)t0[i]; a[base + 4 + i] = (h16)t1[i]; }
}
template <typename T>
__device__ inline void load8g(const T* __restrict__ row, int c, int Kreal, v16h& a, int base) {
#pragma unroll
  for (int i = 0; i < 8; ++i) a[base + i] = (c + i < Kreal) ? (h16)row[c + i] : (h16)0.0f;
}

// ---------------------------------------------------------------------------
// WMMA GEMM:  C[M,Nn] = act( A[M,K] * Bt[Nn,K]^T + bias ) (* rowScale)
// One wave owns a 64x32 output tile = 4(M) x 2(N) WMMA subtiles -> 8
// v_wmma_f32_16x16x32_f16 per 32-wide K step, amortizing fragment loads.
// Requirements (guaranteed by all call sites): M % 64 == 0, Nn % 32 == 0,
// K % 32 == 0 (zero-padded Bt); Kreal guards A's true width (bm: 1000).
// ---------------------------------------------------------------------------
template <typename AT, int ACT, int RS>
__global__ __launch_bounds__(256) void wmma_gemm(
    const AT* __restrict__ A, int lda,
    const h16* __restrict__ Bt, int ldb,
    const float* __restrict__ bias,
    const float* __restrict__ rowScale,
    h16* __restrict__ C, int ldc,
    int M, int Nn, int K, int Kreal)
{
  int wave   = blockIdx.x * (blockDim.x >> 5) + (threadIdx.x >> 5);
  int tilesN = Nn >> 5;                         // 32-wide N strips
  int tilesM = M >> 6;                          // 64-tall M strips
  if (wave >= tilesM * tilesN) return;          // wave-uniform: EXEC stays full
  int tn = wave % tilesN;
  int tm = wave / tilesN;

  int lane   = threadIdx.x & 31;
  int lane16 = lane & 15;
  int hs     = lane >> 4;                       // lane half (CDNA5 A/B K-split)

  const AT* arow[4];
#pragma unroll
  for (int r = 0; r < 4; ++r)
    arow[r] = A + (size_t)(tm * 64 + r * 16 + lane16) * lda;
  const h16* brow[2];
#pragma unroll
  for (int c = 0; c < 2; ++c)
    brow[c] = Bt + (size_t)(tn * 32 + c * 16 + lane16) * ldb;

  v8f acc[4][2] = {};
  for (int k0 = 0; k0 < K; k0 += 32) {
    if (k0 + 64 < K) {                          // uniform: global_prefetch_b8
      __builtin_prefetch(arow[0] + k0 + 64, 0, 1);
      __builtin_prefetch(arow[2] + k0 + 64, 0, 1);
    }
    v16h a[4], b[2];
    // A (16x32, 16-bit): lane-half 0 -> K {k0..+7, k0+16..+23}; half 1 -> +8
    int c0 = k0 + hs * 8;
    int c1 = k0 + 16 + hs * 8;
    if (k0 + 32 <= Kreal) {                     // wave-uniform fast path
#pragma unroll
      for (int r = 0; r < 4; ++r) { load8(arow[r], c0, a[r], 0); load8(arow[r], c1, a[r], 8); }
    } else {                                    // bm tail only (k0=992, Kreal=1000)
#pragma unroll
      for (int r = 0; r < 4; ++r) { load8g(arow[r], c0, Kreal, a[r], 0); load8g(arow[r], c1, Kreal, a[r], 8); }
    }
    // B (32x16): lane = column; half 0 -> K {k0..+7, k0+16..+23}; half 1 -> +8
    int cb = k0 + hs * 8;
#pragma unroll
    for (int c = 0; c < 2; ++c) { load8(brow[c], cb, b[c], 0); load8(brow[c], cb + 16, b[c], 8); }
#pragma unroll
    for (int r = 0; r < 4; ++r)
#pragma unroll
      for (int c = 0; c < 2; ++c)
        acc[r][c] = __builtin_amdgcn_wmma_f32_16x16x32_f16(
            false, a[r], false, b[c], (short)0, acc[r][c], false, false);
  }

  // C/D layout: lanes 0-15 -> M=i, lanes 16-31 -> M=i+8, N = lane16
#pragma unroll
  for (int c = 0; c < 2; ++c) {
    int   n  = tn * 32 + c * 16 + lane16;
    float bv = bias ? bias[n] : 0.0f;
#pragma unroll
    for (int r = 0; r < 4; ++r) {
      int mbase = tm * 64 + r * 16 + hs * 8;
#pragma unroll
      for (int i = 0; i < 8; ++i) {
        int   m = mbase + i;
        float v = acc[r][c][i] + bv;
        if (RS)  v *= rowScale[m];
        if (ACT) v = v > 0.0f ? v : 0.0f;
        C[(size_t)m * ldc + n] = (h16)v;
      }
    }
  }
}

// ---------------------------------------------------------------------------
// Weight transpose + f32->f16: W[K,N] row-major -> Wt[N,Kpad] (zero padded)
// ---------------------------------------------------------------------------
__global__ void convT(const float* __restrict__ W, h16* __restrict__ Wt,
                      int K, int Nn, int Kpad) {
  int idx = blockIdx.x * blockDim.x + threadIdx.x;
  if (idx >= Nn * Kpad) return;
  int n = idx / Kpad, k = idx - n * Kpad;
  Wt[idx] = (k < K) ? (h16)W[(size_t)k * Nn + n] : (h16)0.0f;
}

// ---------------------------------------------------------------------------
// Leaf level: zero dynamic X columns (640..1407) and leaf lrep/rrep carries
// ---------------------------------------------------------------------------
__global__ void leaf_init(h16* __restrict__ X, h16* __restrict__ lrepO,
                          h16* __restrict__ rrepO, int cnt) {
  int idx = blockIdx.x * blockDim.x + threadIdx.x;
  if (idx >= cnt * 768) return;
  int i = idx / 768, c = idx - i * 768;
  X[(size_t)i * XW + 640 + c] = (h16)0.0f;
  if (c < REP) { lrepO[i * REP + c] = (h16)0.0f; rrepO[i * REP + c] = (h16)0.0f; }
}

// ---------------------------------------------------------------------------
// Interior level: masked gather of child reps into X cols 640..1407,
// plus this level's lrep/rrep carries.
// ---------------------------------------------------------------------------
__global__ void build_dyn(const h16* __restrict__ repC, const h16* __restrict__ lrepC,
                          const h16* __restrict__ rrepC,
                          const int* __restrict__ lmask, const int* __restrict__ rmask,
                          h16* __restrict__ X, h16* __restrict__ lrepO,
                          h16* __restrict__ rrepO, int lsh, int cnt, int off) {
  int idx = blockIdx.x * blockDim.x + threadIdx.x;
  if (idx >= cnt * REP) return;
  int i = idx >> 7, h = idx & 127;
  int k  = 1 << lsh;
  int b  = i >> lsh, j = i & (k - 1);
  int cl = ((b << (lsh + 1)) + 2 * j) * REP + h;   // left child row, feature h
  int cr = cl + REP;                               // right child
  h16 lm = (h16)(float)lmask[off + i];
  h16 rm = (h16)(float)rmask[off + i];
  h16 lrep = repC[cl]  * lm, ll = lrepC[cl] * lm, lr = rrepC[cl] * lm;
  h16 rrep = repC[cr]  * rm, rl = lrepC[cr] * rm, rr = rrepC[cr] * rm;
  h16* xr = X + (size_t)i * XW + h;
  xr[640] = lrep; xr[768] = rrep; xr[896] = ll;
  xr[1024] = lr;  xr[1152] = rl;  xr[1280] = rr;
  lrepO[idx] = lrep; rrepO[idx] = rrep;
}

// ---------------------------------------------------------------------------
// AND-pool: min over P=4 predicate embeddings + bias -> X column block
// ---------------------------------------------------------------------------
__global__ void minpool4(const h16* __restrict__ G, const float* __restrict__ bias,
                         h16* __restrict__ X, int colOff, int cnt) {
  int idx = blockIdx.x * blockDim.x + threadIdx.x;
  if (idx >= cnt * HID) return;
  int i = idx >> 7, h = idx & 127;
  const h16* g = G + (size_t)i * 4 * HID + h;
  float m = (float)g[0];
  m = fminf(m, (float)g[HID]);
  m = fminf(m, (float)g[2 * HID]);
  m = fminf(m, (float)g[3 * HID]);
  X[(size_t)i * XW + colOff + h] = (h16)(m + bias[h]);
}

// ---------------------------------------------------------------------------
// Head output: sigmoid( H[i,:] . w + b )    (N=1 matmul -> plain dot)
// ---------------------------------------------------------------------------
__global__ void head_out(const h16* __restrict__ H, const float* __restrict__ w,
                         const float* __restrict__ b, float* __restrict__ out, int M) {
  int i = blockIdx.x * blockDim.x + threadIdx.x;
  if (i >= M) return;
  float s = 0.0f;
  const h16* hr = H + (size_t)i * HID;
#pragma unroll 8
  for (int h = 0; h < HID; ++h) s += (float)hr[h] * w[h];
  out[i] = 1.0f / (1.0f + __expf(-(s + b[0])));
}

// ---------------------------------------------------------------------------
static inline int ceil_div(int a, int b) { return (a + b - 1) / b; }

template <typename AT, int ACT, int RS>
static void launch_gemm(const AT* A, int lda, const h16* Bt, int ldb,
                        const float* bias, const float* rs,
                        h16* C, int ldc, int M, int Nn, int K, int Kreal,
                        hipStream_t s) {
  int tiles  = (M / 64) * (Nn / 32);             // M%64==0, Nn%32==0 at all sites
  int blocks = ceil_div(tiles, 8);               // 8 waves (256 thr) per block
  wmma_gemm<AT, ACT, RS><<<blocks, 256, 0, s>>>(A, lda, Bt, ldb, bias, rs, C, ldc,
                                                M, Nn, K, Kreal);
}

extern "C" void kernel_launch(void* const* d_in, const int* in_sizes, int n_in,
                              void* d_out, int out_size, void* d_ws, size_t ws_size,
                              hipStream_t stream) {
  (void)in_sizes; (void)n_in; (void)out_size; (void)ws_size;

  // ---- inputs (setup_inputs order) ----
  const float* op_vecs   = (const float*)d_in[0];
  const float* feat_vecs = (const float*)d_in[1];
  const float* bitmap    = (const float*)d_in[2];
  const float* pred1     = (const float*)d_in[3];
  const float* pred2     = (const float*)d_in[4];
  const float* has_cond  = (const float*)d_in[5];
  const int*   lmask     = (const int*)d_in[6];
  const int*   rmask     = (const int*)d_in[7];
  const float* W_op  = (const float*)d_in[8],  *b_op  = (const float*)d_in[9];
  const float* W_feat= (const float*)d_in[10], *b_feat= (const float*)d_in[11];
  const float* W_bm  = (const float*)d_in[12], *b_bm  = (const float*)d_in[13];
  const float* W_pred= (const float*)d_in[14], *b_pred= (const float*)d_in[15];
  const float* W_r1  = (const float*)d_in[16], *b_r1  = (const float*)d_in[17];
  const float* W_r2  = (const float*)d_in[18], *b_r2  = (const float*)d_in[19];
  const float* W_r3  = (const float*)d_in[20], *b_r3  = (const float*)d_in[21];
  const float* W_c1  = (const float*)d_in[22], *b_c1  = (const float*)d_in[23];
  const float* W_c2  = (const float*)d_in[24], *b_c2  = (const float*)d_in[25];
  const float* W_c3  = (const float*)d_in[26], *b_c3  = (const float*)d_in[27];
  const float* W_d1  = (const float*)d_in[28], *b_d1  = (const float*)d_in[29];
  const float* W_d2  = (const float*)d_in[30], *b_d2  = (const float*)d_in[31];
  const float* W_d3  = (const float*)d_in[32], *b_d3  = (const float*)d_in[33];
  float* out = (float*)d_out;

  // ---- workspace bump allocator ----
  char* wp = (char*)d_ws;
  auto alloc = [&](size_t halves) -> h16* {
    h16* p = (h16*)wp;
    wp += ((halves * sizeof(h16) + 255) / 256) * 256;
    return p;
  };
  const int MAXCNT = BB << DD;                   // 32768 leaf nodes
  h16* Wt_op   = alloc(HID * 32);
  h16* Wt_feat = alloc(HID * 32);
  h16* Wt_bm   = alloc(HID * BMKP);
  h16* Wt_pred = alloc(HID * 64);
  h16* Wt_r1   = alloc(512 * XW);
  h16* Wt_r2   = alloc(256 * 512);
  h16* Wt_r3   = alloc(HID * 256);
  h16* Wt_c1   = alloc(HID * HID);
  h16* Wt_c2   = alloc(HID * HID);
  h16* Wt_d1   = alloc(HID * HID);
  h16* Wt_d2   = alloc(HID * HID);
  h16* X       = alloc((size_t)MAXCNT * XW);
  h16* G       = alloc((size_t)MAXCNT * 4 * HID);
  h16* H1      = alloc((size_t)MAXCNT * 512);
  h16* H2      = alloc((size_t)MAXCNT * 256);
  h16* repBuf[2]  = { alloc((size_t)MAXCNT * REP), alloc((size_t)MAXCNT * REP) };
  h16* lrepBuf[2] = { alloc((size_t)MAXCNT * REP), alloc((size_t)MAXCNT * REP) };
  h16* rrepBuf[2] = { alloc((size_t)MAXCNT * REP), alloc((size_t)MAXCNT * REP) };
  h16* Hh1     = alloc((size_t)BB * HID);
  h16* Hh2     = alloc((size_t)BB * HID);

  // ---- weight transpose + f16 conversion ----
  auto conv = [&](const float* W, h16* Wt, int K, int Nn, int Kpad) {
    int tot = Nn * Kpad;
    convT<<<ceil_div(tot, 256), 256, 0, stream>>>(W, Wt, K, Nn, Kpad);
  };
  conv(W_op,   Wt_op,   32,   HID, 32);
  conv(W_feat, Wt_feat, 32,   HID, 32);
  conv(W_bm,   Wt_bm,   BMK,  HID, BMKP);
  conv(W_pred, Wt_pred, 64,   HID, 64);
  conv(W_r1,   Wt_r1,   XW,   512, XW);
  conv(W_r2,   Wt_r2,   512,  256, 512);
  conv(W_r3,   Wt_r3,   256,  HID, 256);
  conv(W_c1,   Wt_c1,   HID,  HID, HID);
  conv(W_c2,   Wt_c2,   HID,  HID, HID);
  conv(W_d1,   Wt_d1,   HID,  HID, HID);
  conv(W_d2,   Wt_d2,   HID,  HID, HID);

  // ---- bottom-up level loop ----
  for (int l = DD; l >= 0; --l) {
    int k   = 1 << l;
    int cnt = BB * k;
    int off = BB * (k - 1);
    int pl  = l & 1, pc = (l + 1) & 1;

    if (l == DD) {
      leaf_init<<<ceil_div(cnt * 768, 256), 256, 0, stream>>>(
          X, lrepBuf[pl], rrepBuf[pl], cnt);
    } else {
      build_dyn<<<ceil_div(cnt * REP, 256), 256, 0, stream>>>(
          repBuf[pc], lrepBuf[pc], rrepBuf[pc], lmask, rmask,
          X, lrepBuf[pl], rrepBuf[pl], l, cnt, off);
    }

    // static embedding columns
    launch_gemm<float, 0, 0>(op_vecs   + (size_t)off * 32,  32,   Wt_op,   32,
                             b_op,  nullptr, X + 0,   XW, cnt, HID, 32,  32,  stream);
    launch_gemm<float, 0, 0>(feat_vecs + (size_t)off * 32,  32,   Wt_feat, 32,
                             b_feat, nullptr, X + 128, XW, cnt, HID, 32,  32,  stream);
    launch_gemm<float, 0, 1>(bitmap + (size_t)off * BMK, BMK, Wt_bm, BMKP,
                             b_bm, has_cond + off, X + 512, XW, cnt, HID, BMKP, BMK, stream);
    launch_gemm<float, 0, 0>(pred1 + (size_t)off * 256, 64, Wt_pred, 64,
                             nullptr, nullptr, G, HID, cnt * 4, HID, 64, 64, stream);
    minpool4<<<ceil_div(cnt * HID, 256), 256, 0, stream>>>(G, b_pred, X, 256, cnt);
    launch_gemm<float, 0, 0>(pred2 + (size_t)off * 256, 64, Wt_pred, 64,
                             nullptr, nullptr, G, HID, cnt * 4, HID, 64, 64, stream);
    minpool4<<<ceil_div(cnt * HID, 256), 256, 0, stream>>>(G, b_pred, X, 384, cnt);

    // per-node MLP: 1408 -> 512 -> 256 -> 128, ReLU each
    launch_gemm<h16, 1, 0>(X,  XW,  Wt_r1, XW,  b_r1, nullptr, H1,        512, cnt, 512, XW,  XW,  stream);
    launch_gemm<h16, 1, 0>(H1, 512, Wt_r2, 512, b_r2, nullptr, H2,        256, cnt, 256, 512, 512, stream);
    launch_gemm<h16, 1, 0>(H2, 256, Wt_r3, 256, b_r3, nullptr, repBuf[pl], REP, cnt, REP, 256, 256, stream);
  }

  // ---- heads on root representations (level 0 -> parity 0) ----
  h16* root = repBuf[0];
  launch_gemm<h16, 1, 0>(root, REP, Wt_c1, REP, b_c1, nullptr, Hh1, REP, BB, HID, HID, HID, stream);
  launch_gemm<h16, 1, 0>(Hh1,  REP, Wt_c2, REP, b_c2, nullptr, Hh2, REP, BB, HID, HID, HID, stream);
  head_out<<<ceil_div(BB, 256), 256, 0, stream>>>(Hh2, W_c3, b_c3, out, BB);
  launch_gemm<h16, 1, 0>(root, REP, Wt_d1, REP, b_d1, nullptr, Hh1, REP, BB, HID, HID, HID, stream);
  launch_gemm<h16, 1, 0>(Hh1,  REP, Wt_d2, REP, b_d2, nullptr, Hh2, REP, BB, HID, HID, HID, stream);
  head_out<<<ceil_div(BB, 256), 256, 0, stream>>>(Hh2, W_d3, b_d3, out + BB, BB);
}
